// ResidualMambaBlock_39522289058429
// MI455X (gfx1250) — matrix operand order repaired and checked by reading, not compiled, not run on hardware
//
#include <hip/hip_runtime.h>
#include <hip/hip_bf16.h>
#include <math.h>

// ---------------- problem constants ----------------
#define DIMV      1024
#define D_STATE   128
#define D_CONV    4
#define D_INNER   2048
#define HEADDIM   64
#define NHEADS    32
#define CONV_DIMV (D_INNER + 2 * D_STATE)               // 2304
#define D_IN_PROJ (2 * D_INNER + 2 * D_STATE + NHEADS)  // 4384
#define HIDDENV   4096
#define BATCH     2
#define SEQ       2048
#define ROWS      (BATCH * SEQ)                         // 4096

typedef __attribute__((ext_vector_type(16))) _Float16 v16h;
typedef __attribute__((ext_vector_type(8)))  _Float16 v8h;
typedef __attribute__((ext_vector_type(4)))  _Float16 v4h;
typedef __attribute__((ext_vector_type(8)))  float    v8f;
// GCC-style vector: exact pointee type required by the async-to-LDS builtin
typedef int v4i_vs __attribute__((vector_size(16)));

#define GLOBAL_AS __attribute__((address_space(1)))
#define LDS_AS    __attribute__((address_space(3)))

// gfx1250 async global->LDS staging (ASYNCcnt-tracked), guarded so the
// build can't regress on toolchains that don't declare the builtins.
#if defined(__gfx1250__) && __has_builtin(__builtin_amdgcn_global_load_async_to_lds_b128) && \
    __has_builtin(__builtin_amdgcn_s_wait_asynccnt)
#define USE_ASYNC_LDS 1
#else
#define USE_ASYNC_LDS 0
#endif

__device__ __forceinline__ float sigmoidf_(float x) { return 1.0f / (1.0f + __expf(-x)); }
__device__ __forceinline__ float siluf_(float x)    { return x * sigmoidf_(x); }
__device__ __forceinline__ float softplusf_(float x) {
    return (x > 30.0f) ? x : logf(1.0f + __expf(x));
}

__device__ __forceinline__ v8f wmma16x16x32(v16h a, v16h b, v8f c) {
    // D = A(16x32 f16) * B(32x16 f16) + C(16x16 f32)
    return __builtin_amdgcn_wmma_f32_16x16x32_f16(
        /*neg_a=*/false, a, /*neg_b=*/false, b,
        /*c_mod=*/(short)0, c, /*reuse_a=*/false, /*reuse_b=*/false);
}

__device__ __forceinline__ v16h cat8(v8h lo, v8h hi) {
    return __builtin_shufflevector(lo, hi, 0, 1, 2, 3, 4, 5, 6, 7,
                                            8, 9, 10, 11, 12, 13, 14, 15);
}

// ---------------- f32 -> f16 convert ----------------
__global__ void k_f32_to_f16(const float* __restrict__ in, _Float16* __restrict__ out, int n) {
    int i = blockIdx.x * 256 + threadIdx.x;
    if (i < n) out[i] = (_Float16)in[i];
}

// ---------------- rmsnorm (dim=1024) -> f16 ----------------
__global__ void __launch_bounds__(256)
k_rmsnorm_f16(const float* __restrict__ in, _Float16* __restrict__ out, float eps) {
    int row = blockIdx.x;
    int t = threadIdx.x;
    const float* p = in + (size_t)row * DIMV;
    float v[4];
    float s = 0.0f;
#pragma unroll
    for (int i = 0; i < 4; ++i) { v[i] = p[t + i * 256]; s += v[i] * v[i]; }
#pragma unroll
    for (int off = 16; off > 0; off >>= 1) s += __shfl_xor(s, off, 32);
    __shared__ float warp_s[8];
    __shared__ float scale_s;
    if ((t & 31) == 0) warp_s[t >> 5] = s;
    __syncthreads();
    if (t == 0) {
        float tot = 0.0f;
#pragma unroll
        for (int i = 0; i < 8; ++i) tot += warp_s[i];
        scale_s = rsqrtf(tot / (float)DIMV + eps);
    }
    __syncthreads();
    float sc = scale_s;
    _Float16* o = out + (size_t)row * DIMV;
#pragma unroll
    for (int i = 0; i < 4; ++i) o[t + i * 256] = (_Float16)(v[i] * sc);
}

// ---------------- WMMA GEMM: C[M,N] = A[M,K](f16) * W[N,K](f16)^T ----------------
// block: 256 threads = 8 waves along M; block tile 256(M) x 32(N).
// wave tile 32x32 = 4 WMMAs per 32-wide K step; B staged in double-buffered LDS
// (async global->LDS when available); one barrier per K step.
__global__ void __launch_bounds__(256)
k_gemm_wmma_f16(const _Float16* __restrict__ A, const _Float16* __restrict__ W,
                float* __restrict__ C, int M, int N, int K) {
    __shared__ __align__(16) _Float16 sB[2][32 * 32];   // [buf][n 0..31][k 0..31]

    const int tid  = threadIdx.x;
    const int wid  = tid >> 5;
    const int lane = tid & 31;

    const int mbase = blockIdx.y * 256 + wid * 32;
    const int nbase = blockIdx.x * 32;

    const int arow = lane & 15;           // A row within 16-row fragment
    const int ka0  = (lane >> 4) * 8;     // A K-base per ISA 16-bit A layout
    const int kb0  = (lane >> 4) * 8;     // B K-base, same k order as A
    const int bcol = lane & 15;           // B column within 16-wide sub-tile

    const _Float16* ap0 = A + (size_t)(mbase + arow) * K;        // rows 0..15
    const _Float16* ap1 = ap0 + (size_t)16 * K;                  // rows 16..31

    v8f acc00 = {}, acc01 = {}, acc10 = {}, acc11 = {};

#if USE_ASYNC_LDS
    // 128 lanes x 16B cover the 32x32 f16 tile
    const int srow = tid >> 2;            // 0..63 (only <32 used)
    const int scol = (tid & 3) * 8;       // halfs
#else
    const int srow = tid >> 3;            // 0..31
    const int scol = (tid & 7) * 4;       // halfs
#endif

    // stage one B tile (n-major rows of W, contiguous K) into sB[buf]
    auto stageB = [&](int k, int buf) {
#if USE_ASYNC_LDS
        if (tid < 128) {
            __builtin_amdgcn_global_load_async_to_lds_b128(
                (GLOBAL_AS v4i_vs*)(W + (size_t)(nbase + srow) * K + k + scol),
                (LDS_AS v4i_vs*)(&sB[buf][srow * 32 + scol]), 0, 0);
        }
#else
        *(v4h*)(&sB[buf][srow * 32 + scol]) =
            *(const v4h*)(W + (size_t)(nbase + srow) * K + k + scol);
#endif
    };

    stageB(0, 0);
    int buf = 0;
    for (int k = 0; k < K; k += 32, buf ^= 1) {
#if USE_ASYNC_LDS
        __builtin_amdgcn_s_wait_asynccnt(0);
#endif
        __syncthreads();                        // tile[buf] visible to all waves
        if (k + 32 < K) {
            stageB(k + 32, buf ^ 1);            // overlap next stage with compute
            __builtin_prefetch(ap0 + k + 32, 0, 3);
            __builtin_prefetch(ap1 + k + 32, 0, 3);
        }

        // A fragments: lane holds A[row][ka0..+7] and A[row][ka0+16..+23]
        v16h a0 = cat8(*(const v8h*)(ap0 + k + ka0), *(const v8h*)(ap0 + k + ka0 + 16));
        v16h a1 = cat8(*(const v8h*)(ap1 + k + ka0), *(const v8h*)(ap1 + k + ka0 + 16));

        // B fragments: lane holds W[nbase+col][k+kb0 .. +7] and [k+kb0+16 .. +23]
        const _Float16* bb = &sB[buf][0];
        v16h b0 = cat8(*(const v8h*)(bb + bcol * 32 + kb0),
                       *(const v8h*)(bb + bcol * 32 + kb0 + 16));
        v16h b1 = cat8(*(const v8h*)(bb + (16 + bcol) * 32 + kb0),
                       *(const v8h*)(bb + (16 + bcol) * 32 + kb0 + 16));

        acc00 = wmma16x16x32(a0, b0, acc00);
        acc01 = wmma16x16x32(a0, b1, acc01);
        acc10 = wmma16x16x32(a1, b0, acc10);
        acc11 = wmma16x16x32(a1, b1, acc11);
    }

    // C layout: VGPR r -> row (r + 8*(lane>>4)), col = lane&15
    const int roff = 8 * (lane >> 4);
    const int ccol = nbase + (lane & 15);
    float* cp0 = C + (size_t)(mbase + roff) * N + ccol;
    float* cp1 = C + (size_t)(mbase + 16 + roff) * N + ccol;
#pragma unroll
    for (int r = 0; r < 8; ++r) {
        cp0[(size_t)r * N]      = acc00[r];
        cp0[(size_t)r * N + 16] = acc01[r];
        cp1[(size_t)r * N]      = acc10[r];
        cp1[(size_t)r * N + 16] = acc11[r];
    }
}

// ---------------- depthwise causal conv (k=4) + bias + SiLU ----------------
__global__ void k_conv_silu(const float* __restrict__ zx, const float* __restrict__ convw,
                            const float* __restrict__ convb, float* __restrict__ xbc) {
    int idx = blockIdx.x * 256 + threadIdx.x;
    if (idx >= BATCH * SEQ * CONV_DIMV) return;
    int c = idx % CONV_DIMV;
    int l = (idx / CONV_DIMV) % SEQ;
    int b = idx / (CONV_DIMV * SEQ);
    float acc = convb[c];
#pragma unroll
    for (int j = 0; j < D_CONV; ++j) {
        int ll = l - (D_CONV - 1) + j;
        if (ll >= 0)
            acc += convw[c * D_CONV + j] *
                   zx[((size_t)b * SEQ + ll) * D_IN_PROJ + D_INNER + c];
    }
    xbc[idx] = siluf_(acc);
}

// ---------------- sequential SSM scan: one block per (batch, head) ----------------
// state h[HEADDIM][D_STATE] held in registers: 256 threads x 32 f32
__global__ void __launch_bounds__(256)
k_scan(const float* __restrict__ xbc, const float* __restrict__ zx,
       const float* __restrict__ dt_bias, const float* __restrict__ A_log,
       const float* __restrict__ Dvec, float* __restrict__ y) {
    const int blk  = blockIdx.x;          // 0..63
    const int b    = blk / NHEADS;
    const int head = blk % NHEADS;
    const int t    = threadIdx.x;
    const int p    = t >> 2;              // 0..63  (HEADDIM)
    const int nq   = t & 3;               // 4 quadrants of D_STATE
    const int n0   = nq * 32;

    __shared__ float sx[HEADDIM];
    __shared__ float sB[D_STATE];
    __shared__ float sC[D_STATE];
    __shared__ float sdt[2];              // {dt, dA}

    const float Ac  = -__expf(A_log[head]);
    const float dtb = dt_bias[head];
    const float Dh  = Dvec[head];

    float h[32];
#pragma unroll
    for (int i = 0; i < 32; ++i) h[i] = 0.0f;

    for (int step = 0; step < SEQ; ++step) {
        const float* rowp = xbc + ((size_t)b * SEQ + step) * CONV_DIMV;
        for (int i = t; i < HEADDIM + 2 * D_STATE; i += 256) {
            if (i < HEADDIM)                sx[i] = rowp[head * HEADDIM + i];
            else if (i < HEADDIM + D_STATE) sB[i - HEADDIM] = rowp[D_INNER + (i - HEADDIM)];
            else sC[i - HEADDIM - D_STATE] = rowp[D_INNER + D_STATE + (i - HEADDIM - D_STATE)];
        }
        if (t == 0) {
            float raw = zx[((size_t)b * SEQ + step) * D_IN_PROJ + (D_IN_PROJ - NHEADS) + head] + dtb;
            float dtv = softplusf_(raw);
            sdt[0] = dtv;
            sdt[1] = __expf(dtv * Ac);
        }
        __syncthreads();

        const float xp  = sx[p];
        const float dtx = sdt[0] * xp;
        const float dA  = sdt[1];
        float ysum = 0.0f;
#pragma unroll
        for (int i = 0; i < 32; ++i) {
            const int n = n0 + i;
            h[i] = dA * h[i] + dtx * sB[n];
            ysum += h[i] * sC[n];
        }
        // reduce over the 4 threads sharing one p (adjacent lanes in a wave32)
        ysum += __shfl_xor(ysum, 1, 32);
        ysum += __shfl_xor(ysum, 2, 32);
        if (nq == 0)
            y[((size_t)b * SEQ + step) * D_INNER + head * HEADDIM + p] = ysum + Dh * xp;
        __syncthreads();
    }
}

// ---------------- y * silu(z), rmsnorm(eps=1e-5), * norm_w -> f16 (dim=2048) ----------------
__global__ void __launch_bounds__(256)
k_gate_norm(const float* __restrict__ y, const float* __restrict__ zx,
            const float* __restrict__ norm_w, _Float16* __restrict__ out) {
    int row = blockIdx.x;
    int t = threadIdx.x;
    const float* yp = y + (size_t)row * D_INNER;
    const float* zp = zx + (size_t)row * D_IN_PROJ;   // z = first D_INNER cols
    float v[8];
    float s = 0.0f;
#pragma unroll
    for (int i = 0; i < 8; ++i) {
        int idx = t + i * 256;
        float zv = zp[idx];
        float val = yp[idx] * siluf_(zv);
        v[i] = val;
        s += val * val;
    }
#pragma unroll
    for (int off = 16; off > 0; off >>= 1) s += __shfl_xor(s, off, 32);
    __shared__ float warp_s[8];
    __shared__ float scale_s;
    if ((t & 31) == 0) warp_s[t >> 5] = s;
    __syncthreads();
    if (t == 0) {
        float tot = 0.0f;
#pragma unroll
        for (int i = 0; i < 8; ++i) tot += warp_s[i];
        scale_s = rsqrtf(tot / (float)D_INNER + 1e-5f);
    }
    __syncthreads();
    float sc = scale_s;
    _Float16* o = out + (size_t)row * D_INNER;
#pragma unroll
    for (int i = 0; i < 8; ++i) {
        int idx = t + i * 256;
        o[idx] = (_Float16)(v[i] * sc * norm_w[idx]);
    }
}

// ---------------- x1 = x + scale*mo ; h16 = f16(rmsnorm(x1)) (dim=1024) ----------------
__global__ void __launch_bounds__(256)
k_resid_rmsnorm(const float* __restrict__ x, const float* __restrict__ mo,
                const float* __restrict__ svec, float* __restrict__ x1,
                _Float16* __restrict__ h16, float eps) {
    int row = blockIdx.x;
    int t = threadIdx.x;
    const float* xp = x + (size_t)row * DIMV;
    const float* mp = mo + (size_t)row * DIMV;
    float* x1p = x1 + (size_t)row * DIMV;
    float v[4];
    float s = 0.0f;
#pragma unroll
    for (int i = 0; i < 4; ++i) {
        int idx = t + i * 256;
        float val = xp[idx] + svec[idx] * mp[idx];
        v[i] = val;
        x1p[idx] = val;
        s += val * val;
    }
#pragma unroll
    for (int off = 16; off > 0; off >>= 1) s += __shfl_xor(s, off, 32);
    __shared__ float warp_s[8];
    __shared__ float scale_s;
    if ((t & 31) == 0) warp_s[t >> 5] = s;
    __syncthreads();
    if (t == 0) {
        float tot = 0.0f;
#pragma unroll
        for (int i = 0; i < 8; ++i) tot += warp_s[i];
        scale_s = rsqrtf(tot / (float)DIMV + eps);
    }
    __syncthreads();
    float sc = scale_s;
    _Float16* o = h16 + (size_t)row * DIMV;
#pragma unroll
    for (int i = 0; i < 4; ++i) o[t + i * 256] = (_Float16)(v[i] * sc);
}

// ---------------- act = f16( silu(g) * u ) ----------------
__global__ void k_silu_mul_f16(const float* __restrict__ g, const float* __restrict__ u,
                               _Float16* __restrict__ out, int n) {
    int i = blockIdx.x * 256 + threadIdx.x;
    if (i < n) out[i] = (_Float16)(siluf_(g[i]) * u[i]);
}

// ---------------- out = x1 + mlp_scale * mlp ----------------
__global__ void k_final_add(const float* __restrict__ x1, const float* __restrict__ mlp,
                            const float* __restrict__ svec, float* __restrict__ out, int n) {
    int i = blockIdx.x * 256 + threadIdx.x;
    if (i < n) out[i] = x1[i] + svec[i % DIMV] * mlp[i];
}

// =====================================================================
extern "C" void kernel_launch(void* const* d_in, const int* in_sizes, int n_in,
                              void* d_out, int out_size, void* d_ws, size_t ws_size,
                              hipStream_t stream) {
    const float* x           = (const float*)d_in[0];
    const float* in_proj_w   = (const float*)d_in[1];
    const float* conv_w      = (const float*)d_in[2];
    const float* conv_b      = (const float*)d_in[3];
    const float* dt_bias     = (const float*)d_in[4];
    const float* A_log       = (const float*)d_in[5];
    const float* Dvec        = (const float*)d_in[6];
    const float* norm_w      = (const float*)d_in[7];
    const float* out_proj_w  = (const float*)d_in[8];
    const float* gate_w      = (const float*)d_in[9];
    const float* up_w        = (const float*)d_in[10];
    const float* down_w      = (const float*)d_in[11];
    const float* mixer_scale = (const float*)d_in[12];
    const float* mlp_scale   = (const float*)d_in[13];
    float* out = (float*)d_out;

    // ---- workspace bump allocator ----
    char* ws = (char*)d_ws;
    size_t off = 0;
    auto alloc = [&](size_t bytes) -> char* {
        char* p = ws + off;
        off += (bytes + 255) & ~(size_t)255;
        return p;
    };
    _Float16* u16   = (_Float16*)alloc((size_t)ROWS * DIMV * 2);
    _Float16* win16 = (_Float16*)alloc((size_t)D_IN_PROJ * DIMV * 2);
    _Float16* wout16= (_Float16*)alloc((size_t)DIMV * D_INNER * 2);
    _Float16* wg16  = (_Float16*)alloc((size_t)HIDDENV * DIMV * 2);
    _Float16* wu16  = (_Float16*)alloc((size_t)HIDDENV * DIMV * 2);
    _Float16* wd16  = (_Float16*)alloc((size_t)DIMV * HIDDENV * 2);
    float*    zx    = (float*)   alloc((size_t)ROWS * D_IN_PROJ * 4);
    float*    xbc   = (float*)   alloc((size_t)ROWS * CONV_DIMV * 4);
    float*    ybuf  = (float*)   alloc((size_t)ROWS * D_INNER * 4);
    _Float16* yb16  = (_Float16*)alloc((size_t)ROWS * D_INNER * 2);
    float*    mo    = (float*)   alloc((size_t)ROWS * DIMV * 4);
    float*    x1    = (float*)   alloc((size_t)ROWS * DIMV * 4);
    _Float16* h16   = (_Float16*)alloc((size_t)ROWS * DIMV * 2);
    float*    gbuf  = (float*)   alloc((size_t)ROWS * HIDDENV * 4);
    float*    ubuf  = (float*)   alloc((size_t)ROWS * HIDDENV * 4);
    _Float16* act16 = (_Float16*)alloc((size_t)ROWS * HIDDENV * 2);
    float*    mlp   = (float*)   alloc((size_t)ROWS * DIMV * 4);
    (void)ws_size; (void)in_sizes; (void)n_in; (void)out_size;

    const float eps_f32 = 1.1920929e-07f;  // jnp.finfo(float32).eps

    auto cvt = [&](const float* src, _Float16* dst, size_t n) {
        k_f32_to_f16<<<(unsigned)((n + 255) / 256), 256, 0, stream>>>(src, dst, (int)n);
    };

    // 1) weight conversions to f16
    cvt(in_proj_w,  win16, (size_t)D_IN_PROJ * DIMV);
    cvt(out_proj_w, wout16,(size_t)DIMV * D_INNER);
    cvt(gate_w,     wg16,  (size_t)HIDDENV * DIMV);
    cvt(up_w,       wu16,  (size_t)HIDDENV * DIMV);
    cvt(down_w,     wd16,  (size_t)DIMV * HIDDENV);

    // 2) u = rmsnorm(x) -> f16
    k_rmsnorm_f16<<<ROWS, 256, 0, stream>>>(x, u16, eps_f32);

    // 3) zxbcdt = u @ in_proj_w^T   (M=4096, N=4384, K=1024)
    k_gemm_wmma_f16<<<dim3(D_IN_PROJ / 32, ROWS / 256), 256, 0, stream>>>(
        u16, win16, zx, ROWS, D_IN_PROJ, DIMV);

    // 4) depthwise causal conv + SiLU
    {
        int n = BATCH * SEQ * CONV_DIMV;
        k_conv_silu<<<(n + 255) / 256, 256, 0, stream>>>(zx, conv_w, conv_b, xbc);
    }

    // 5) sequential SSM scan (one block per (batch, head); state in VGPRs)
    k_scan<<<BATCH * NHEADS, 256, 0, stream>>>(xbc, zx, dt_bias, A_log, Dvec, ybuf);

    // 6) gated rmsnorm -> f16
    k_gate_norm<<<ROWS, 256, 0, stream>>>(ybuf, zx, norm_w, yb16);

    // 7) mamba_out = ybar @ out_proj_w^T   (M=4096, N=1024, K=2048)
    k_gemm_wmma_f16<<<dim3(DIMV / 32, ROWS / 256), 256, 0, stream>>>(
        yb16, wout16, mo, ROWS, DIMV, D_INNER);

    // 8) x1 = x + mixer_scale*mo ; h16 = rmsnorm(x1)
    k_resid_rmsnorm<<<ROWS, 256, 0, stream>>>(x, mo, mixer_scale, x1, h16, eps_f32);

    // 9) gate / up GEMMs  (M=4096, N=4096, K=1024)
    k_gemm_wmma_f16<<<dim3(HIDDENV / 32, ROWS / 256), 256, 0, stream>>>(
        h16, wg16, gbuf, ROWS, HIDDENV, DIMV);
    k_gemm_wmma_f16<<<dim3(HIDDENV / 32, ROWS / 256), 256, 0, stream>>>(
        h16, wu16, ubuf, ROWS, HIDDENV, DIMV);

    // 10) act = silu(g)*u -> f16
    {
        int n = ROWS * HIDDENV;
        k_silu_mul_f16<<<(n + 255) / 256, 256, 0, stream>>>(gbuf, ubuf, act16, n);
    }

    // 11) mlp = act @ down_w^T   (M=4096, N=1024, K=4096)
    k_gemm_wmma_f16<<<dim3(DIMV / 32, ROWS / 256), 256, 0, stream>>>(
        act16, wd16, mlp, ROWS, DIMV, HIDDENV);

    // 12) out = x1 + mlp_scale*mlp
    {
        int n = ROWS * DIMV;
        k_final_add<<<(n + 255) / 256, 256, 0, stream>>>(x1, mlp, mlp_scale, out, n);
    }
}
